// coAttention_alt_10582799417982
// MI455X (gfx1250) — compile-verified, matrix operand order checked by reading, not run on hardware
//
#include <hip/hip_runtime.h>

// ===========================================================================
// MI455X (gfx1250) fused co-attention.
//
// Key algebraic fact: in _attend, the broadcast query contributes a constant
// (per batch) shift to every logit, and softmax is shift-invariant => the
// whole q0/q1/q2 projection chain (and the feature means feeding it) cannot
// affect the output. Output == Vi + Vt + 0.5*Vc + 0.5*Ve with four fully
// independent attention stages.
//
// Roofline: 30.5 GFLOP of proj GEMMs vs ~240 MB of mandatory feature reads
// => memory bound at 23.3 TB/s (~10 us). Strategy: read each feature tensor
// once (f32 global -> f16 LDS), run the proj GEMM with v_wmma_f32_16x16x32_f16
// out of LDS, reuse the same LDS copy for the P*feat contraction. Weights are
// pre-transposed/converted to f16 (2 MB total) and stay L2-resident across all
// 256 blocks.
//
// CDNA5 paths used: v_wmma_f32_16x16x32_f16 (8 waves x 2 k-tiles, persistent
// f32 accumulators), global_load_async_to_lds_b128 double-buffered weight
// slab (ASYNCcnt + s_wait_asynccnt), native v_tanh_f32 when available,
// split s_barrier.
// ===========================================================================

typedef _Float16 v4h  __attribute__((ext_vector_type(4)));
typedef _Float16 v8h  __attribute__((ext_vector_type(8)));
typedef _Float16 v16h __attribute__((ext_vector_type(16)));
typedef float    v8f  __attribute__((ext_vector_type(8)));
typedef int      v4i  __attribute__((ext_vector_type(4)));

#define D_DIM 1024
#define K_DIM 256
#define B_DIM 256
#define FSTRIDE 1032   // feat LDS row stride (halves): +8 pad -> conflict-free A tiles
#define WSTRIDE 40     // W slab LDS row stride (halves): +8 pad -> conflict-free B tiles

#if __has_builtin(__builtin_amdgcn_global_load_async_to_lds_b128) && \
    __has_builtin(__builtin_amdgcn_s_wait_asynccnt)
#define HAVE_ASYNC_LDS 1
#else
#define HAVE_ASYNC_LDS 0
#endif

#if HAVE_ASYNC_LDS
typedef v4i __attribute__((address_space(1))) g_v4i;   // global 128-bit chunk
typedef v4i __attribute__((address_space(3))) l_v4i;   // LDS    128-bit chunk
#endif

// ---------------------------------------------------------------------------
// W (D,K) f32  ->  Wt (K,D) f16   (tiled transpose, 4 matrices)
// ---------------------------------------------------------------------------
__global__ __launch_bounds__(256) void wt_cvt_kernel(
    const float* __restrict__ w0, const float* __restrict__ w1,
    const float* __restrict__ w2, const float* __restrict__ w3,
    _Float16* __restrict__ wt)
{
  __shared__ float tile[32][33];
  const float* W = (blockIdx.z == 0) ? w0 : (blockIdx.z == 1) ? w1
                 : (blockIdx.z == 2) ? w2 : w3;
  _Float16* Wt = wt + (size_t)blockIdx.z * K_DIM * D_DIM;
  const int dB = blockIdx.x * 32, kB = blockIdx.y * 32;
  const int tx = threadIdx.x, ty = threadIdx.y;
#pragma unroll
  for (int r = 0; r < 32; r += 8)
    tile[ty + r][tx] = W[(size_t)(dB + ty + r) * K_DIM + (kB + tx)];
  __syncthreads();
#pragma unroll
  for (int r = 0; r < 32; r += 8)
    Wt[(size_t)(kB + ty + r) * D_DIM + (dB + tx)] = (_Float16)tile[tx][ty + r];
}

// Load one 16x32 f16 WMMA operand slice for this lane:
// 8 contiguous halves at p, 8 more at p+16 (the half-wave K interleave).
__device__ __forceinline__ v16h load_tile16(const _Float16* p) {
  v8h lo = *(const v8h*)p;
  v8h hi = *(const v8h*)(p + 16);
  return __builtin_shufflevector(lo, hi, 0, 1, 2, 3, 4, 5, 6, 7,
                                         8, 9, 10, 11, 12, 13, 14, 15);
}

// Copy one 64-byte Wt row into the LDS slab. On CDNA5 use the async
// global->LDS DMA path (ASYNCcnt) so the copy bypasses VGPRs and overlaps
// with the WMMA stream; fall back to load+ds_store otherwise.
__device__ __forceinline__ void stage_slab_row(const _Float16* __restrict__ src,
                                               _Float16* dst) {
#if HAVE_ASYNC_LDS
  g_v4i* g = (g_v4i*)(unsigned long long)src;
  l_v4i* l = (l_v4i*)(unsigned int)(unsigned long long)dst;
  __builtin_amdgcn_global_load_async_to_lds_b128(g,     l,     0, 0);
  __builtin_amdgcn_global_load_async_to_lds_b128(g + 1, l + 1, 0, 0);
  __builtin_amdgcn_global_load_async_to_lds_b128(g + 2, l + 2, 0, 0);
  __builtin_amdgcn_global_load_async_to_lds_b128(g + 3, l + 3, 0, 0);
#else
  const v8h* s = (const v8h*)src;
  v8h* d = (v8h*)dst;
  d[0] = s[0]; d[1] = s[1]; d[2] = s[2]; d[3] = s[3];
#endif
}

__device__ __forceinline__ void async_join() {
#if HAVE_ASYNC_LDS
  __builtin_amdgcn_s_wait_asynccnt(0);
#endif
}

__device__ __forceinline__ float fast_tanh(float x) {
#if __has_builtin(__builtin_amdgcn_tanhf)
  return __builtin_amdgcn_tanhf(x);          // v_tanh_f32
#elif __has_builtin(__builtin_amdgcn_tanh_f32)
  return __builtin_amdgcn_tanh_f32(x);
#else
  return tanhf(x);
#endif
}

// ---------------------------------------------------------------------------
// Fused attend: one block per batch element, 8 waves.
//   proj = feat @ W (WMMA f16->f32), logits = sum_k tanh(proj)*wp2 + bias,
//   P = softmax(logits), out (+)= scale * P @ feat
// NT = number of 16-row n-tiles (N padded with zero rows).
// ---------------------------------------------------------------------------
template <int NT, bool ACCUM>
__global__ __launch_bounds__(256) void attend_kernel(
    const float* __restrict__ feat, int N,
    const _Float16* __restrict__ Wt,     // (K, D) f16
    const float* __restrict__ wp2,       // second half of wp, length K
    const float* __restrict__ bias,      // length N
    float* __restrict__ out, float scale)
{
  extern __shared__ char smem[];
  _Float16* feath = (_Float16*)smem;                          // NT*16 x FSTRIDE
  _Float16* wslab = feath + NT * 16 * FSTRIDE;                // 2 x K x WSTRIDE
  float* part   = (float*)(wslab + 2 * K_DIM * WSTRIDE);      // [8][NT*16]
  float* logits = part + 8 * NT * 16;                         // [NT*16]
  float* red    = logits + NT * 16;                           // [2]

  const int tid  = threadIdx.x;
  const int lane = tid & 31;
  const int wave = tid >> 5;
  const int hh   = lane >> 4;   // half-wave
  const int l16  = lane & 15;
  const int b    = blockIdx.x;

  // ---- stage first Wt slab (d0 = 0) via async DMA while we convert feat ----
  stage_slab_row(Wt + (size_t)tid * D_DIM, wslab + tid * WSTRIDE);

  // ---- stage feat[b] (N x D f32) -> f16 LDS; zero-pad rows N..NT*16-1 ----
  {
    const float4* src = (const float4*)(feat + (size_t)b * N * D_DIM);
    const int n4 = N * (D_DIM / 4);
    for (int i = tid; i < n4; i += 256) {
      const int row = i >> 8;           // D_DIM/4 == 256
      const int c4  = i & 255;
      float4 v = src[i];
      v4h h = { (_Float16)v.x, (_Float16)v.y, (_Float16)v.z, (_Float16)v.w };
      *(v4h*)(feath + (size_t)row * FSTRIDE + c4 * 4) = h;
    }
    for (int i = n4 + tid; i < NT * 16 * (D_DIM / 4); i += 256) {
      const int row = i >> 8;
      const int c4  = i & 255;
      v4h z = {};
      *(v4h*)(feath + (size_t)row * FSTRIDE + c4 * 4) = z;
    }
  }

  // wave w owns k-tiles {2w, 2w+1}; preload its wp2 weights
  const int kt0 = wave * 2, kt1 = wave * 2 + 1;
  const float wpa = wp2[kt0 * 16 + l16];
  const float wpb = wp2[kt1 * 16 + l16];

  v8f acc[NT][2];
#pragma unroll
  for (int nt = 0; nt < NT; ++nt) {
    v8f z = {0.f, 0.f, 0.f, 0.f, 0.f, 0.f, 0.f, 0.f};
    acc[nt][0] = z;
    acc[nt][1] = z;
  }

  async_join();
  __syncthreads();

  const int off = hh ? 8 : 0;
#pragma unroll 1
  for (int step = 0; step < D_DIM / 32; ++step) {
    const int cur = step & 1;
    // prefetch next 32-wide d-slab of Wt into the other LDS buffer (async DMA
    // overlaps with the WMMA work below until the asynccnt join at loop end)
    if (step + 1 < D_DIM / 32) {
      stage_slab_row(Wt + (size_t)tid * D_DIM + (step + 1) * 32,
                     wslab + (cur ^ 1) * K_DIM * WSTRIDE + tid * WSTRIDE);
    }
    const _Float16* wb = wslab + cur * K_DIM * WSTRIDE;
    v16h b0 = load_tile16(wb + (kt0 * 16 + l16) * WSTRIDE + off);
    v16h b1 = load_tile16(wb + (kt1 * 16 + l16) * WSTRIDE + off);
    const int d0 = step * 32;
#pragma unroll
    for (int nt = 0; nt < NT; ++nt) {
      v16h a = load_tile16(feath + (size_t)(nt * 16 + l16) * FSTRIDE + d0 + off);
      acc[nt][0] = __builtin_amdgcn_wmma_f32_16x16x32_f16(
          false, a, false, b0, (short)0, acc[nt][0], false, false);
      acc[nt][1] = __builtin_amdgcn_wmma_f32_16x16x32_f16(
          false, a, false, b1, (short)0, acc[nt][1], false, false);
    }
    async_join();
    __syncthreads();
  }

  // ---- logits[n] = sum_k tanh(proj[n,k]) * wp2[k]  (deterministic) ----
#pragma unroll
  for (int nt = 0; nt < NT; ++nt) {
#pragma unroll
    for (int r = 0; r < 8; ++r) {
      float t = fast_tanh(acc[nt][0][r]) * wpa + fast_tanh(acc[nt][1][r]) * wpb;
      t += __shfl_xor(t, 1, 32);
      t += __shfl_xor(t, 2, 32);
      t += __shfl_xor(t, 4, 32);
      t += __shfl_xor(t, 8, 32);
      if (l16 == 0)
        part[wave * (NT * 16) + nt * 16 + hh * 8 + r] = t;
    }
  }
  __syncthreads();

  if (tid < NT * 16) {
    float s = 0.f;
#pragma unroll
    for (int w = 0; w < 8; ++w) s += part[w * (NT * 16) + tid];
    logits[tid] = (tid < N) ? (s + bias[tid]) : -1e30f;
  }
  __syncthreads();
  if (tid == 0) {
    float m = -1e30f;
    for (int n = 0; n < N; ++n) m = fmaxf(m, logits[n]);
    red[0] = m;
  }
  __syncthreads();
  if (tid < N) logits[tid] = __expf(logits[tid] - red[0]);
  __syncthreads();
  if (tid == 0) {
    float s = 0.f;
    for (int n = 0; n < N; ++n) s += logits[n];
    red[1] = 1.f / s;
  }
  __syncthreads();
  if (tid < N) logits[tid] *= red[1];
  __syncthreads();

  // ---- out[b, d] (+)= scale * sum_n P[n] * feat[n, d] ----
  const int d = tid * 4;
  float a0 = 0.f, a1 = 0.f, a2 = 0.f, a3 = 0.f;
  for (int n = 0; n < N; ++n) {
    const float p = logits[n];
    v4h f = *(const v4h*)(feath + (size_t)n * FSTRIDE + d);
    a0 += p * (float)f[0];
    a1 += p * (float)f[1];
    a2 += p * (float)f[2];
    a3 += p * (float)f[3];
  }
  float4* o4 = (float4*)(out + (size_t)b * D_DIM + d);
  float4 res;
  res.x = scale * a0; res.y = scale * a1; res.z = scale * a2; res.w = scale * a3;
  if (ACCUM) {
    float4 old = *o4;
    res.x += old.x; res.y += old.y; res.z += old.z; res.w += old.w;
  }
  *o4 = res;
}

static size_t attend_smem(int NT) {
  return (size_t)NT * 16 * FSTRIDE * sizeof(_Float16)
       + 2 * (size_t)K_DIM * WSTRIDE * sizeof(_Float16)
       + (size_t)(8 * NT * 16 + NT * 16 + 2) * sizeof(float);
}

extern "C" void kernel_launch(void* const* d_in, const int* in_sizes, int n_in,
                              void* d_out, int out_size, void* d_ws, size_t ws_size,
                              hipStream_t stream) {
  (void)in_sizes; (void)n_in; (void)out_size; (void)ws_size;

  const float* ifeat_h = (const float*)d_in[0];   // (B, 49, D)
  const float* tfeat_h = (const float*)d_in[1];   // (B, 128, D)
  const float* cfeat   = (const float*)d_in[2];   // (B, 20, D)
  const float* efeat   = (const float*)d_in[3];   // (B, 30, D)
  // d_in[4] (w_Vit_0), d_in[12] (w_Vtce_1), d_in[15] (w_Vice_2) feed only the
  // q-path, which is softmax-shift-invariant dead code.
  const float* w_Vc = (const float*)d_in[5];
  const float* wp_c = (const float*)d_in[6];
  const float* b_c  = (const float*)d_in[7];
  const float* w_Ve = (const float*)d_in[8];
  const float* wp_e = (const float*)d_in[9];
  const float* b_e  = (const float*)d_in[10];
  const float* w_Vi = (const float*)d_in[11];
  const float* wp_i = (const float*)d_in[13];
  const float* b_i  = (const float*)d_in[14];
  const float* w_Vt = (const float*)d_in[16];
  const float* wp_t = (const float*)d_in[17];
  const float* b_t  = (const float*)d_in[18];

  float*     out = (float*)d_out;
  _Float16*  wt  = (_Float16*)d_ws;   // 4 * K * D f16 = 2 MB (L2-resident)

  // allow >64KB dynamic LDS (WGP has 320KB)
  (void)hipFuncSetAttribute(reinterpret_cast<const void*>(&attend_kernel<2, false>),
                      hipFuncAttributeMaxDynamicSharedMemorySize, (int)attend_smem(2));
  (void)hipFuncSetAttribute(reinterpret_cast<const void*>(&attend_kernel<2, true>),
                      hipFuncAttributeMaxDynamicSharedMemorySize, (int)attend_smem(2));
  (void)hipFuncSetAttribute(reinterpret_cast<const void*>(&attend_kernel<4, true>),
                      hipFuncAttributeMaxDynamicSharedMemorySize, (int)attend_smem(4));
  (void)hipFuncSetAttribute(reinterpret_cast<const void*>(&attend_kernel<8, true>),
                      hipFuncAttributeMaxDynamicSharedMemorySize, (int)attend_smem(8));

  // weights -> f16, transposed to (K, D)
  wt_cvt_kernel<<<dim3(D_DIM / 32, K_DIM / 32, 4), dim3(32, 8), 0, stream>>>(
      w_Vc, w_Ve, w_Vi, w_Vt, wt);

  const size_t WSZ = (size_t)K_DIM * D_DIM;
  // out  = 0.5 * Vc
  attend_kernel<2, false><<<B_DIM, 256, attend_smem(2), stream>>>(
      cfeat, 20, wt + 0 * WSZ, wp_c + K_DIM, b_c, out, 0.5f);
  // out += 0.5 * Ve
  attend_kernel<2, true><<<B_DIM, 256, attend_smem(2), stream>>>(
      efeat, 30, wt + 1 * WSZ, wp_e + K_DIM, b_e, out, 0.5f);
  // out += Vi
  attend_kernel<4, true><<<B_DIM, 256, attend_smem(4), stream>>>(
      ifeat_h, 49, wt + 2 * WSZ, wp_i + K_DIM, b_i, out, 1.0f);
  // out += Vt
  attend_kernel<8, true><<<B_DIM, 256, attend_smem(8), stream>>>(
      tfeat_h, 128, wt + 3 * WSZ, wp_t + K_DIM, b_t, out, 1.0f);
}
